// Model_83519934038720
// MI455X (gfx1250) — matrix operand order repaired and checked by reading, not compiled, not run on hardware
//
#include <hip/hip_runtime.h>

#define N_NODES 100000
#define N_EDGES 800000
#define IN_CH   128
#define HID     64
#define N_ITERS 15
#define LDS_LD  72   // padded row stride for 16x64 LDS tiles

typedef __attribute__((ext_vector_type(16))) __bf16 v16bf;
typedef __attribute__((ext_vector_type(8)))  float  v8f;

// D = A*B + C with bf16 hi/lo split: A ~ Ahi + Alo, B ~ Bhi + Blo.
// Keep hi*hi + hi*lo + lo*hi (drop lo*lo): ~f32 accuracy on the bf16 WMMA pipe.
__device__ __forceinline__ v8f wmma_hl(v16bf ahi, v16bf alo, v16bf bhi, v16bf blo, v8f c) {
  c = __builtin_amdgcn_wmma_f32_16x16x32_bf16(false, alo, false, bhi, (short)0, c, false, false);
  c = __builtin_amdgcn_wmma_f32_16x16x32_bf16(false, ahi, false, blo, (short)0, c, false, false);
  c = __builtin_amdgcn_wmma_f32_16x16x32_bf16(false, ahi, false, bhi, (short)0, c, false, false);
  return c;
}

// A fragment (16x32 bf16): lane<16 -> row M=lane, K bases {0..7,16..23};
// lanes 16-31 -> same rows, K offset +8. RELU optionally fused into the load.
template <bool RELU>
__device__ __forceinline__ void load_a(const float* A, int lda, int k0, int lane,
                                       v16bf& hi, v16bf& lo) {
  const int row  = lane & 15;
  const int half = lane >> 4;
  const float* base = A + (size_t)row * lda + k0 + half * 8;
  #pragma unroll
  for (int j = 0; j < 16; ++j) {
    const int v = j >> 1;
    const int k = ((v & 4) << 2) + ((v & 3) << 1) + (j & 1);  // {0..7, 16..23}
    float x = base[k];
    if (RELU) x = fmaxf(x, 0.0f);
    const __bf16 h = (__bf16)x;
    hi[j] = h;
    lo[j] = (__bf16)(x - (float)h);
  }
}

// One 16xHID output tile. A is 16xK (global or LDS); B comes from PREPPED
// fragment buffers (hi/lo bf16, WMMA lane order): frag[(k0/32)*128 + n*32 + lane].
template <bool RELU>
__device__ __forceinline__ void gemm_tile(const float* A, int lda,
                                          const v16bf* __restrict__ WH,
                                          const v16bf* __restrict__ WL,
                                          int K, int lane, v8f acc[4]) {
  for (int k0 = 0, blk = 0; k0 < K; k0 += 32, ++blk) {
    v16bf ahi, alo;
    load_a<RELU>(A, lda, k0, lane, ahi, alo);
    #pragma unroll
    for (int n = 0; n < 4; ++n) {
      const int idx = blk * 128 + n * 32 + lane;
      acc[n] = wmma_hl(ahi, alo, WH[idx], WL[idx], acc[n]);
    }
  }
}

// C layout: VGPR v, lanes 0-15 -> row v, lanes 16-31 -> row v+8; N = n*16 + lane%16.
__device__ __forceinline__ void store_c(float* dst, int ldd, const v8f acc[4], int lane,
                                        bool dotanh) {
  const int half = lane >> 4;
  const int colb = lane & 15;
  #pragma unroll
  for (int n = 0; n < 4; ++n) {
    #pragma unroll
    for (int v = 0; v < 8; ++v) {
      float val = acc[n][v];
      if (dotanh) val = tanhf(val);
      dst[(size_t)(v + half * 8) * ldd + n * 16 + colb] = val;
    }
  }
}

// Non-returning hardware f32 atomic add (STOREcnt-tracked, no CAS loop).
__device__ __forceinline__ void atom_add_f32(float* p, float v) {
  asm volatile("global_atomic_add_f32 %0, %1, off" : : "v"(p), "v"(v) : "memory");
}

// ---- Kernel 0: pre-split a KxHID f32 weight into bf16 hi/lo WMMA fragments ----
// One thread per fragment: tid -> (k-block, n-tile, lane); 16 bf16 per lane, 32B store.
__global__ __launch_bounds__(256) void k_prep(const float* __restrict__ W, int K,
                                              v16bf* __restrict__ H, v16bf* __restrict__ L) {
  const int tid   = (int)blockIdx.x * 256 + (int)threadIdx.x;
  const int nfrag = (K >> 5) * 128;
  if (tid >= nfrag) return;
  const int blk   = tid >> 7;
  const int n     = (tid >> 5) & 3;
  const int lane  = tid & 31;
  const int col   = n * 16 + (lane & 15);
  const int kbase = blk * 32 + (lane >> 4) * 8;    // lanes 0-15: K={0..7,16..23}, 16-31: K={8..15,24..31}
  v16bf hi, lo;
  #pragma unroll
  for (int j = 0; j < 16; ++j) {
    const float x = W[(size_t)(kbase + j + ((j & 8) ? 8 : 0)) * HID + col];
    const __bf16 h = (__bf16)x;
    hi[j] = h;
    lo[j] = (__bf16)(x - (float)h);
  }
  H[tid] = hi;
  L[tid] = lo;
}

// ---- Kernel 1: fused encoder + bias MLP:  b = tanh((x@W_enc)@Wb1)@Wb2 ----
__global__ __launch_bounds__(256) void k_encode_bias(
    const float* __restrict__ x,
    const v16bf* __restrict__ wencH, const v16bf* __restrict__ wencL,
    const v16bf* __restrict__ wb1H,  const v16bf* __restrict__ wb1L,
    const v16bf* __restrict__ wb2H,  const v16bf* __restrict__ wb2L,
    float* __restrict__ b_out) {
  __shared__ float tile[8][16 * LDS_LD];
  const int lane = threadIdx.x & 31;
  const int wave = threadIdx.x >> 5;
  const int row0 = ((int)blockIdx.x * 8 + wave) * 16;
  if (row0 >= N_NODES) return;
  float* t = tile[wave];

  v8f acc[4] = {v8f{}, v8f{}, v8f{}, v8f{}};
  gemm_tile<false>(x + (size_t)row0 * IN_CH, IN_CH, wencH, wencL, IN_CH, lane, acc);  // h
  store_c(t, LDS_LD, acc, lane, false);
  asm volatile("s_wait_dscnt 0" ::: "memory");

  v8f acc2[4] = {v8f{}, v8f{}, v8f{}, v8f{}};
  gemm_tile<false>(t, LDS_LD, wb1H, wb1L, HID, lane, acc2);                           // h@Wb1
  store_c(t, LDS_LD, acc2, lane, true);                                               // tanh -> LDS
  asm volatile("s_wait_dscnt 0" ::: "memory");

  v8f acc3[4] = {v8f{}, v8f{}, v8f{}, v8f{}};
  gemm_tile<false>(t, LDS_LD, wb2H, wb2L, HID, lane, acc3);                           // b
  store_c(b_out + (size_t)row0 * HID, HID, acc3, lane, false);
}

// ---- Kernel 2: init  u=0, u_half=-b, acc=0 ----
__global__ __launch_bounds__(256) void k_init(
    const float* __restrict__ b, float* __restrict__ u,
    float* __restrict__ uh, float* __restrict__ acc, int total4) {
  const int i = (int)blockIdx.x * (int)blockDim.x + (int)threadIdx.x;
  if (i >= total4) return;
  const float4 bb = ((const float4*)b)[i];
  const float4 z  = make_float4(0.f, 0.f, 0.f, 0.f);
  ((float4*)u)[i]   = z;
  ((float4*)acc)[i] = z;
  ((float4*)uh)[i]  = make_float4(-bb.x, -bb.y, -bb.z, -bb.w);
}

// ---- Kernel 3: sparse propagation  acc[dst] += w * u_half[src] ----
// Wave-per-edge; lane l covers channels {2l, 2l+1} -> coalesced 256B gather,
// hardware global_atomic_add_f32 into the 192MB-L2-resident accumulator.
__global__ __launch_bounds__(256) void k_scatter(
    const long long* __restrict__ ei, const float* __restrict__ ew,
    const float* __restrict__ uh, float* __restrict__ acc) {
  const int lane  = threadIdx.x & 31;
  const int gwave = (int)(((size_t)blockIdx.x * blockDim.x + threadIdx.x) >> 5);
  const int nwave = (int)(((size_t)gridDim.x * blockDim.x) >> 5);
  for (int e = gwave; e < N_EDGES; e += nwave) {
    const int   src = (int)ei[e];
    const int   dst = (int)ei[N_EDGES + e];
    const float w   = ew[e];
    const float2 v  = ((const float2*)(uh + (size_t)src * HID))[lane];
    float* ap = acc + (size_t)dst * HID + 2 * lane;
    atom_add_f32(ap,     w * v.x);
    atom_add_f32(ap + 1, w * v.y);
  }
}

// ---- Kernel 4: fused PR update + next-iteration prep ----
// un = 2*acc - 2*relu(u) + u ; u := un ; if !last: uh = 2*relu(un)-un-b, acc := 0
__global__ __launch_bounds__(256) void k_update(
    float* __restrict__ u, float* __restrict__ acc,
    const float* __restrict__ b, float* __restrict__ uh,
    int total4, int last) {
  const int i = (int)blockIdx.x * (int)blockDim.x + (int)threadIdx.x;
  if (i >= total4) return;
  const float4 a  = ((const float4*)acc)[i];
  const float4 uo = ((const float4*)u)[i];
  float4 un;
  un.x = 2.f * a.x - 2.f * fmaxf(uo.x, 0.f) + uo.x;
  un.y = 2.f * a.y - 2.f * fmaxf(uo.y, 0.f) + uo.y;
  un.z = 2.f * a.z - 2.f * fmaxf(uo.z, 0.f) + uo.z;
  un.w = 2.f * a.w - 2.f * fmaxf(uo.w, 0.f) + uo.w;
  ((float4*)u)[i] = un;
  if (!last) {
    const float4 bb = ((const float4*)b)[i];
    float4 h;
    h.x = 2.f * fmaxf(un.x, 0.f) - un.x - bb.x;
    h.y = 2.f * fmaxf(un.y, 0.f) - un.y - bb.y;
    h.z = 2.f * fmaxf(un.z, 0.f) - un.z - bb.z;
    h.w = 2.f * fmaxf(un.w, 0.f) - un.w - bb.w;
    ((float4*)uh)[i]  = h;
    ((float4*)acc)[i] = make_float4(0.f, 0.f, 0.f, 0.f);
  }
}

// ---- Kernel 5: decoder  out = relu(u) @ W_dec ----
__global__ __launch_bounds__(256) void k_decode(
    const float* __restrict__ u,
    const v16bf* __restrict__ wdecH, const v16bf* __restrict__ wdecL,
    float* __restrict__ out) {
  const int lane = threadIdx.x & 31;
  const int wave = threadIdx.x >> 5;
  const int row0 = ((int)blockIdx.x * 8 + wave) * 16;
  if (row0 >= N_NODES) return;
  v8f acc[4] = {v8f{}, v8f{}, v8f{}, v8f{}};
  gemm_tile<true>(u + (size_t)row0 * HID, HID, wdecH, wdecL, HID, lane, acc);  // relu fused
  store_c(out + (size_t)row0 * HID, HID, acc, lane, false);
}

extern "C" void kernel_launch(void* const* d_in, const int* in_sizes, int n_in,
                              void* d_out, int out_size, void* d_ws, size_t ws_size,
                              hipStream_t stream) {
  (void)in_sizes; (void)n_in; (void)out_size; (void)ws_size;
  const float*     x     = (const float*)d_in[0];
  const long long* ei    = (const long long*)d_in[1];  // int64 [2, E]
  const float*     ew    = (const float*)d_in[2];
  const float*     W_enc = (const float*)d_in[3];
  const float*     Wb1   = (const float*)d_in[4];
  const float*     Wb2   = (const float*)d_in[5];
  const float*     W_dec = (const float*)d_in[6];
  float*           out   = (float*)d_out;

  // Workspace: 4 node-feature buffers (25.6 MB each) + weight fragment buffers (80 KB).
  float* ws  = (float*)d_ws;
  const size_t per = (size_t)N_NODES * HID;
  float* b   = ws;
  float* u   = ws + per;
  float* uh  = ws + 2 * per;
  float* acc = ws + 3 * per;
  v16bf* fr    = (v16bf*)(ws + 4 * per);   // 32B-aligned (offset is a multiple of 32)
  v16bf* wencH = fr;                       // (128/32)*128 = 512 frags
  v16bf* wencL = wencH + 512;
  v16bf* wb1H  = wencL + 512;              // (64/32)*128 = 256 frags each below
  v16bf* wb1L  = wb1H + 256;
  v16bf* wb2H  = wb1L + 256;
  v16bf* wb2L  = wb2H + 256;
  v16bf* wdecH = wb2L + 256;
  v16bf* wdecL = wdecH + 256;

  const dim3 blk(256);
  const int tile_blocks = (N_NODES + 127) / 128;        // 8 waves x 16 rows per block
  const int total4      = (int)(per / 4);
  const int ew_blocks   = (total4 + 255) / 256;

  k_prep<<<2, blk, 0, stream>>>(W_enc, IN_CH, wencH, wencL);
  k_prep<<<1, blk, 0, stream>>>(Wb1,   HID,   wb1H,  wb1L);
  k_prep<<<1, blk, 0, stream>>>(Wb2,   HID,   wb2H,  wb2L);
  k_prep<<<1, blk, 0, stream>>>(W_dec, HID,   wdecH, wdecL);

  k_encode_bias<<<tile_blocks, blk, 0, stream>>>(x, wencH, wencL, wb1H, wb1L, wb2H, wb2L, b);
  k_init<<<ew_blocks, blk, 0, stream>>>(b, u, uh, acc, total4);
  for (int it = 0; it < N_ITERS; ++it) {
    k_scatter<<<4096, blk, 0, stream>>>(ei, ew, uh, acc);
    k_update<<<ew_blocks, blk, 0, stream>>>(u, acc, b, uh, total4, it == N_ITERS - 1 ? 1 : 0);
  }
  k_decode<<<tile_blocks, blk, 0, stream>>>(u, wdecH, wdecL, out);
}
